// Pointnet2SSG_72739566125560
// MI455X (gfx1250) — matrix operand order repaired and checked by reading, not compiled, not run on hardware
//
#include <hip/hip_runtime.h>

typedef __attribute__((ext_vector_type(16))) _Float16 v16h;
typedef __attribute__((ext_vector_type(8)))  float    v8f;

#define BN_RSQRT 0.9999950000374997f   /* 1/sqrt(1+1e-5) */

// ---------------------------------------------------------------- utilities

// pointcloud (B*N,6) -> xyz (B*N,3) f32 and feats0 (B*N x 3) f16 (point-major)
__global__ void k_split(const float* __restrict__ pc, float* __restrict__ xyz,
                        _Float16* __restrict__ f0, int total) {
  int t = blockIdx.x * blockDim.x + threadIdx.x;
  if (t >= total) return;
  const float* s = pc + (size_t)t * 6;
  xyz[(size_t)t * 3 + 0] = s[0];
  xyz[(size_t)t * 3 + 1] = s[1];
  xyz[(size_t)t * 3 + 2] = s[2];
  f0[(size_t)t * 3 + 0] = (_Float16)s[3];
  f0[(size_t)t * 3 + 1] = (_Float16)s[4];
  f0[(size_t)t * 3 + 2] = (_Float16)s[5];
}

// pack W (Cout x Cin f32) -> (Cout x Kpad f16), zero-padded in K
__global__ void k_packw(const float* __restrict__ W, int Cout, int Cin, int Kpad,
                        _Float16* __restrict__ Wp) {
  int t = blockIdx.x * blockDim.x + threadIdx.x;
  if (t >= Cout * Kpad) return;
  int k = t % Kpad, co = t / Kpad;
  Wp[t] = (_Float16)((k < Cin) ? W[(size_t)co * Cin + k] : 0.0f);
}

// ---------------------------------------------------------------- FPS (serial per batch)

#define FPS_T 256
__global__ void k_fps(const float* __restrict__ xyz, int N, int npoint,
                      int* __restrict__ fidx, float* __restrict__ dists) {
  int b = blockIdx.x;
  const float* p = xyz + (size_t)b * N * 3;
  float* dd = dists + (size_t)b * N;
  __shared__ float sv[FPS_T];
  __shared__ int   si[FPS_T];
  __shared__ int   s_last;
  for (int i = threadIdx.x; i < N; i += FPS_T) dd[i] = 1e10f;
  if (threadIdx.x == 0) { fidx[(size_t)b * npoint] = 0; s_last = 0; }
  __syncthreads();
  for (int it = 1; it < npoint; ++it) {
    int last = s_last;
    float lx = p[last * 3], ly = p[last * 3 + 1], lz = p[last * 3 + 2];
    float best = -1.0f; int bi = 0;
    for (int i = threadIdx.x; i < N; i += FPS_T) {
      float dx = p[i * 3] - lx, dy = p[i * 3 + 1] - ly, dz = p[i * 3 + 2] - lz;
      float d = dx * dx + dy * dy + dz * dz;
      float dm = fminf(dd[i], d);
      dd[i] = dm;
      if (dm > best) { best = dm; bi = i; }
    }
    sv[threadIdx.x] = best; si[threadIdx.x] = bi;
    __syncthreads();
    for (int s = FPS_T / 2; s > 0; s >>= 1) {
      if (threadIdx.x < s) {
        if (sv[threadIdx.x + s] > sv[threadIdx.x] ||
            (sv[threadIdx.x + s] == sv[threadIdx.x] && si[threadIdx.x + s] < si[threadIdx.x])) {
          sv[threadIdx.x] = sv[threadIdx.x + s];
          si[threadIdx.x] = si[threadIdx.x + s];
        }
      }
      __syncthreads();
    }
    if (threadIdx.x == 0) { fidx[(size_t)b * npoint + it] = si[0]; s_last = si[0]; }
    __syncthreads();
  }
}

__global__ void k_newxyz(const float* __restrict__ xyz, const int* __restrict__ fidx,
                         float* __restrict__ nx, int Np, int Nin, int total) {
  int t = blockIdx.x * blockDim.x + threadIdx.x;
  if (t >= total) return;
  int b = t / Np;
  int j = fidx[t];
  const float* s = xyz + ((size_t)b * Nin + j) * 3;
  nx[(size_t)t * 3 + 0] = s[0];
  nx[(size_t)t * 3 + 1] = s[1];
  nx[(size_t)t * 3 + 2] = s[2];
}

// ---------------------------------------------------------------- ball query (first ns in-radius indices, pad with first)

__global__ void k_ballquery(const float* __restrict__ xyz, const float* __restrict__ nxyz,
                            int* __restrict__ gidx, int N, int Np, int ns, float r2, int total) {
  int t = blockIdx.x * blockDim.x + threadIdx.x;
  if (t >= total) return;
  int b = t / Np;
  const float* p = xyz + (size_t)b * N * 3;
  float cx = nxyz[(size_t)t * 3], cy = nxyz[(size_t)t * 3 + 1], cz = nxyz[(size_t)t * 3 + 2];
  int* out = gidx + (size_t)t * ns;
  int cnt = 0, first = 0;
  for (int j = 0; j < N && cnt < ns; ++j) {
    float dx = p[j * 3] - cx, dy = p[j * 3 + 1] - cy, dz = p[j * 3 + 2] - cz;
    if (dx * dx + dy * dy + dz * dz <= r2) {
      if (cnt == 0) first = j;
      out[cnt++] = j;
    }
  }
  for (; cnt < ns; ++cnt) out[cnt] = first;
}

// grouped GEMM input X (M x Kpad, point-major) = [xyz-delta ; gathered feats ; zero-pad]
__global__ void k_group(const float* __restrict__ xyz, const float* __restrict__ nxyz,
                        const int* __restrict__ gidx, const _Float16* __restrict__ fin,
                        int Cprev, int Nin, int Np, int ns,
                        _Float16* __restrict__ X, int Kpad, int M) {
  int t = blockIdx.x * blockDim.x + threadIdx.x;
  if (t >= M) return;
  int m = (t / ns) % Np;
  int b = t / (ns * Np);
  int j = gidx[t];
  const float* p = xyz + ((size_t)b * Nin + j) * 3;
  const float* c = nxyz + ((size_t)b * Np + m) * 3;
  _Float16* xr = X + (size_t)t * Kpad;
  xr[0] = (_Float16)(p[0] - c[0]);
  xr[1] = (_Float16)(p[1] - c[1]);
  xr[2] = (_Float16)(p[2] - c[2]);
  const _Float16* fr = fin + ((size_t)b * Nin + j) * Cprev;
  for (int cc = 0; cc < Cprev; ++cc) xr[3 + cc] = fr[cc];
  for (int k = 3 + Cprev; k < Kpad; ++k) xr[k] = (_Float16)0.0f;
}

// ---------------------------------------------------------------- WMMA GEMM
// Y(m, co) = relu(g/sqrt(1+eps) * sum_k W(co,k) X(m,k) + b)
// X point-major (M x ldx), Y point-major (M x ldy). Each wave: 16 rows x 64 cols,
// one A fragment reused against 4 B fragments -> 4 v_wmma per K step.

__device__ __forceinline__ void store_tile(const v8f& acc, _Float16* yr,
                                           const float* sc, const float* bs, int relu) {
#pragma unroll
  for (int r = 0; r < 8; ++r) {
    float v = acc[r] * sc[r] + bs[r];
    if (relu) v = fmaxf(v, 0.0f);
    yr[r] = (_Float16)v;
  }
}

__global__ void k_gemm_wmma(const _Float16* __restrict__ Wp,   // Cout x Kpad, f16
                            const _Float16* __restrict__ X, int ldx,
                            const float* __restrict__ gvec,
                            const float* __restrict__ bvec,
                            _Float16* __restrict__ Y, int ldy,
                            int M, int Kpad, int relu) {
  const int lane = threadIdx.x & 31;
  const int wave = threadIdx.x >> 5;
  const int m0 = (blockIdx.x * 4 + wave) * 64;
  const int co0 = blockIdx.y * 16;
  if (m0 >= M) return;                        // wave-uniform -> EXEC all-1s for WMMA
  const int row = lane & 15;
  const int seg = lane >> 4;
  const _Float16* wr = Wp + (size_t)(co0 + row) * Kpad + seg * 16;
  const _Float16* x0 = X + (size_t)(m0 + row) * ldx + seg * 16;   // contiguous K per lane
  const _Float16* x1 = x0 + (size_t)16 * ldx;
  const _Float16* x2 = x0 + (size_t)32 * ldx;
  const _Float16* x3 = x0 + (size_t)48 * ldx;
  v8f acc0 = {}, acc1 = {}, acc2 = {}, acc3 = {};
  for (int kt = 0; kt < Kpad; kt += 32) {
    v16h a, b0, b1, b2, b3;
#pragma unroll
    for (int i = 0; i < 8; ++i) {             // A layout: 16 consecutive K per lane-half
      a[i]     = wr[kt + i];
      a[8 + i] = wr[kt + 8 + i];
    }
#pragma unroll
    for (int e = 0; e < 16; ++e) {            // B layout: 16 consecutive K per lane-half
      b0[e] = x0[kt + e];
      b1[e] = x1[kt + e];
      b2[e] = x2[kt + e];
      b3[e] = x3[kt + e];
    }
    __builtin_prefetch(x0 + kt + 32, 0, 1);
    acc0 = __builtin_amdgcn_wmma_f32_16x16x32_f16(false, a, false, b0, (short)0, acc0, false, false);
    acc1 = __builtin_amdgcn_wmma_f32_16x16x32_f16(false, a, false, b1, (short)0, acc1, false, false);
    acc2 = __builtin_amdgcn_wmma_f32_16x16x32_f16(false, a, false, b2, (short)0, acc2, false, false);
    acc3 = __builtin_amdgcn_wmma_f32_16x16x32_f16(false, a, false, b3, (short)0, acc3, false, false);
  }
  float sc[8], bs[8];
#pragma unroll
  for (int r = 0; r < 8; ++r) {               // D layout: VGPR r -> row co0 + 8*seg + r
    int co = co0 + seg * 8 + r;
    sc[r] = gvec[co] * BN_RSQRT;
    bs[r] = bvec[co];
  }
  size_t cobase = (size_t)co0 + seg * 8;
  store_tile(acc0, Y + (size_t)(m0 +  0 + row) * ldy + cobase, sc, bs, relu);
  store_tile(acc1, Y + (size_t)(m0 + 16 + row) * ldy + cobase, sc, bs, relu);
  store_tile(acc2, Y + (size_t)(m0 + 32 + row) * ldy + cobase, sc, bs, relu);
  store_tile(acc3, Y + (size_t)(m0 + 48 + row) * ldy + cobase, sc, bs, relu);
}

// max over ns neighbor samples: Y (B*Np*ns x C) -> F (B*Np x C), point-major
__global__ void k_maxpool(const _Float16* __restrict__ Y, int C, int ns, int cols,
                          _Float16* __restrict__ F) {
  int t = blockIdx.x * blockDim.x + threadIdx.x;
  if (t >= cols * C) return;
  int c = t % C;
  int col = t / C;
  const _Float16* yr = Y + ((size_t)col * ns) * C + c;
  float mx = -1e30f;
  for (int s = 0; s < ns; ++s) mx = fmaxf(mx, (float)yr[(size_t)s * C]);
  F[(size_t)col * C + c] = (_Float16)mx;
}

// ---------------------------------------------------------------- feature propagation: 3-NN interp + skip concat into X

__global__ void k_fp_interp(const float* __restrict__ uxyz, const float* __restrict__ kxyz,
                            const _Float16* __restrict__ kf, int Ck,
                            const _Float16* __restrict__ skipf, int Cs,
                            int Nu, int Nk, int B, _Float16* __restrict__ X, int Kpad) {
  int t = blockIdx.x * blockDim.x + threadIdx.x;
  if (t >= B * Nu) return;
  int b = t / Nu;
  const float* kz = kxyz + (size_t)b * Nk * 3;
  float px = uxyz[(size_t)t * 3], py = uxyz[(size_t)t * 3 + 1], pz = uxyz[(size_t)t * 3 + 2];
  float d0 = 1e30f, d1 = 1e30f, d2 = 1e30f;
  int i0 = 0, i1 = 0, i2 = 0;
  for (int j = 0; j < Nk; ++j) {
    float dx = kz[j * 3] - px, dy = kz[j * 3 + 1] - py, dz = kz[j * 3 + 2] - pz;
    float d = dx * dx + dy * dy + dz * dz;
    if (d < d0)      { d2 = d1; i2 = i1; d1 = d0; i1 = i0; d0 = d; i0 = j; }
    else if (d < d1) { d2 = d1; i2 = i1; d1 = d;  i1 = j; }
    else if (d < d2) { d2 = d;  i2 = j; }
  }
  float w0 = 1.0f / (d0 + 1e-8f), w1 = 1.0f / (d1 + 1e-8f), w2 = 1.0f / (d2 + 1e-8f);
  float wsum = w0 + w1 + w2;
  w0 /= wsum; w1 /= wsum; w2 /= wsum;
  _Float16* xr = X + (size_t)t * Kpad;
  const _Float16* f0p = kf + ((size_t)b * Nk + i0) * Ck;
  const _Float16* f1p = kf + ((size_t)b * Nk + i1) * Ck;
  const _Float16* f2p = kf + ((size_t)b * Nk + i2) * Ck;
  for (int c = 0; c < Ck; ++c)
    xr[c] = (_Float16)(w0 * (float)f0p[c] + w1 * (float)f1p[c] + w2 * (float)f2p[c]);
  const _Float16* sp = skipf + (size_t)t * Cs;
  for (int c = 0; c < Cs; ++c) xr[Ck + c] = sp[c];
  for (int k = Ck + Cs; k < Kpad; ++k) xr[k] = (_Float16)0.0f;
}

// final 3-channel linear head (no BN/relu), f32 output in (B,N,3) order
__global__ void k_fc2(const _Float16* __restrict__ X, const float* __restrict__ W2,
                      const float* __restrict__ b2, float* __restrict__ out, int M) {
  int t = blockIdx.x * blockDim.x + threadIdx.x;
  if (t >= M) return;
  const _Float16* xr = X + (size_t)t * 128;
  float a0 = b2[0], a1 = b2[1], a2 = b2[2];
  for (int c = 0; c < 128; ++c) {
    float x = (float)xr[c];
    a0 += W2[c] * x;
    a1 += W2[128 + c] * x;
    a2 += W2[256 + c] * x;
  }
  out[(size_t)t * 3 + 0] = a0;
  out[(size_t)t * 3 + 1] = a1;
  out[(size_t)t * 3 + 2] = a2;
}

// ---------------------------------------------------------------- host orchestration

struct GL { const float *W, *g, *b; int cin, cout, kpad; _Float16* wp; };

static inline int rup32(int x) { return (x + 31) & ~31; }
static inline unsigned blkn(size_t n, int t) { return (unsigned)((n + t - 1) / t); }

extern "C" void kernel_launch(void* const* d_in, const int* in_sizes, int n_in,
                              void* d_out, int out_size, void* d_ws, size_t ws_size,
                              hipStream_t stream) {
  (void)in_sizes; (void)n_in; (void)out_size; (void)ws_size;
  const int B = 8, N0 = 16384;
  const int Nl[5]      = {16384, 1024, 256, 64, 16};
  const int npts[4]    = {1024, 256, 64, 16};
  const float radii[4] = {0.1f, 0.2f, 0.4f, 0.8f};
  const int NS = 32;
  static const int sa_dims[4][4] = {{6,32,32,64},{67,64,64,128},{131,128,128,256},{259,256,256,512}};
  static const int fp_dims[4][4] = {{131,128,128,128},{320,256,128,0},{384,256,256,0},{768,256,256,0}};
  static const int fp_nl[4] = {3, 2, 2, 2};

  const float* pc = (const float*)d_in[0];

  // ---- parameter walk (dict insertion order: pointcloud, sa, fp, fc1, fc2)
  GL gl[22];
  int idx = 1, gi = 0;
  for (int l = 0; l < 4; ++l)
    for (int j = 0; j < 3; ++j) {
      gl[gi].W = (const float*)d_in[idx++];
      gl[gi].g = (const float*)d_in[idx++];
      gl[gi].b = (const float*)d_in[idx++];
      gl[gi].cin = sa_dims[l][j]; gl[gi].cout = sa_dims[l][j + 1];
      gl[gi].kpad = rup32(gl[gi].cin); ++gi;
    }
  int fpBase[4];
  for (int l = 0; l < 4; ++l) {
    fpBase[l] = gi;
    for (int j = 0; j < fp_nl[l]; ++j) {
      gl[gi].W = (const float*)d_in[idx++];
      gl[gi].g = (const float*)d_in[idx++];
      gl[gi].b = (const float*)d_in[idx++];
      gl[gi].cin = fp_dims[l][j]; gl[gi].cout = fp_dims[l][j + 1];
      gl[gi].kpad = rup32(gl[gi].cin); ++gi;
    }
  }
  int fc1i = gi;
  gl[gi].W = (const float*)d_in[idx++];
  gl[gi].g = (const float*)d_in[idx++];
  gl[gi].b = (const float*)d_in[idx++];
  gl[gi].cin = 128; gl[gi].cout = 128; gl[gi].kpad = 128; ++gi;
  const float* fc2W = (const float*)d_in[idx++];
  const float* fc2b = (const float*)d_in[idx++];

  // ---- workspace carve
  char* wsp = (char*)d_ws;
  size_t off = 0;
  auto alloc = [&](size_t bytes) -> char* {
    char* p = wsp + off;
    off = (off + bytes + 255) & ~(size_t)255;
    return p;
  };
  float*    xyz0 = (float*)alloc((size_t)B * N0 * 3 * 4);
  _Float16* f0   = (_Float16*)alloc((size_t)B * N0 * 3 * 2);
  float*    nx[4];
  for (int l = 0; l < 4; ++l) nx[l] = (float*)alloc((size_t)B * npts[l] * 3 * 4);
  int*      fidx  = (int*)alloc((size_t)B * 1024 * 4);
  int*      gidx  = (int*)alloc((size_t)B * 1024 * NS * 4);
  float*    dists = (float*)alloc((size_t)B * N0 * 4);
  for (int g = 0; g < gi; ++g) gl[g].wp = (_Float16*)alloc((size_t)gl[g].cout * gl[g].kpad * 2);
  _Float16* Xb = (_Float16*)alloc((size_t)160 * 131072 * 2);   // max M*Kpad (FP level-0)
  _Float16* YA = (_Float16*)alloc((size_t)64 * 262144 * 2);    // max M*Cout
  _Float16* YB = (_Float16*)alloc((size_t)64 * 262144 * 2);
  _Float16* featsL[5];
  featsL[1] = (_Float16*)alloc((size_t)B * 1024 * 128 * 2);
  featsL[2] = (_Float16*)alloc((size_t)B * 256 * 256 * 2);
  featsL[3] = (_Float16*)alloc((size_t)B * 64 * 256 * 2);
  featsL[4] = (_Float16*)alloc((size_t)B * 16 * 512 * 2);

  // ---- pack all weights to f16
  for (int g = 0; g < gi; ++g) {
    int tot = gl[g].cout * gl[g].kpad;
    k_packw<<<blkn(tot, 256), 256, 0, stream>>>(gl[g].W, gl[g].cout, gl[g].cin, gl[g].kpad, gl[g].wp);
  }

  auto launch_gemm = [&](const GL& L, const _Float16* Xin, _Float16* Yout, int M) {
    dim3 grid((unsigned)((M / 64 + 3) / 4), (unsigned)(L.cout / 16));
    k_gemm_wmma<<<grid, 128, 0, stream>>>(L.wp, Xin, L.kpad, L.g, L.b, Yout, L.cout, M, L.kpad, 1);
  };

  // ---- level 0 split
  k_split<<<blkn((size_t)B * N0, 256), 256, 0, stream>>>(pc, xyz0, f0, B * N0);

  float* lx[5]; _Float16* lf[5]; int lc[5];
  lx[0] = xyz0; lf[0] = f0; lc[0] = 3;

  // ---- SA stages
  float* curX = xyz0; _Float16* curF = f0;
  int curN = N0, curC = 3;
  for (int l = 0; l < 4; ++l) {
    int Np = npts[l];
    int tot = B * Np;
    k_fps<<<B, FPS_T, 0, stream>>>(curX, curN, Np, fidx, dists);
    k_newxyz<<<blkn(tot, 256), 256, 0, stream>>>(curX, fidx, nx[l], Np, curN, tot);
    float r2 = radii[l] * radii[l];
    k_ballquery<<<blkn(tot, 256), 256, 0, stream>>>(curX, nx[l], gidx, curN, Np, NS, r2, tot);
    int M = B * Np * NS;
    int Cin = 3 + curC, Kpad = rup32(Cin);
    k_group<<<blkn(M, 256), 256, 0, stream>>>(curX, nx[l], gidx, curF, curC, curN, Np, NS, Xb, Kpad, M);
    const _Float16* in = Xb;
    for (int j = 0; j < 3; ++j) {
      _Float16* out = (j == 1) ? YB : YA;
      launch_gemm(gl[l * 3 + j], in, out, M);
      in = out;
    }
    int Cl = sa_dims[l][3];
    k_maxpool<<<blkn((size_t)Cl * tot, 256), 256, 0, stream>>>(in, Cl, NS, tot, featsL[l + 1]);
    curX = nx[l]; curN = Np; curF = featsL[l + 1]; curC = Cl;
    lx[l + 1] = nx[l]; lf[l + 1] = featsL[l + 1]; lc[l + 1] = Cl;
  }

  // ---- FP stages (known level k -> unknown level u)
  for (int s = 0; s < 4; ++s) {
    int u = 3 - s, k = 4 - s;
    int fpi = 3 - s;                       // params['fp'][-1-s]
    int Ck = lc[k], Cs = lc[u];
    int Nu = Nl[u], Nk = Nl[k];
    int M = B * Nu;
    int Kpad = rup32(Ck + Cs);
    k_fp_interp<<<blkn(M, 256), 256, 0, stream>>>(lx[u], lx[k], lf[k], Ck, lf[u], Cs, Nu, Nk, B, Xb, Kpad);
    int nlay = fp_nl[fpi];
    const _Float16* in = Xb;
    _Float16* lastOut = (u > 0) ? featsL[u] : YA;
    for (int j = 0; j < nlay; ++j) {
      _Float16* out = (j == nlay - 1) ? lastOut : ((j == 0) ? YA : YB);
      launch_gemm(gl[fpBase[fpi] + j], in, out, M);
      in = out;
    }
    lf[u] = lastOut;
    lc[u] = fp_dims[fpi][nlay];
  }

  // ---- head: fc1 (WMMA GEMM + BN + relu) then fc2 -> d_out (B,N,3) f32
  int M0 = B * N0;
  launch_gemm(gl[fc1i], lf[0], YB, M0);
  k_fc2<<<blkn(M0, 256), 256, 0, stream>>>(YB, fc2W, fc2b, (float*)d_out, M0);
}
